// AttentionHead_17008070492724
// MI455X (gfx1250) — matrix operand order repaired and checked by reading, not compiled, not run on hardware
//
#include <hip/hip_runtime.h>

#define N_EMBD 2048
#define HEADD  128
#define SEQ    4096
#define BATCH  4
#define QKV3   384   // 3*HEADD

typedef __attribute__((ext_vector_type(16))) __bf16 v16bf;
typedef __attribute__((ext_vector_type(8)))  float  v8f;
typedef __attribute__((ext_vector_type(4)))  unsigned int u32x4;
typedef __attribute__((ext_vector_type(8)))  int i32x8;
typedef __attribute__((ext_vector_type(4)))  int i32x4;

// Tensor Data Mover availability / arity (bridge doc: ROCm7.2 = 5 args,
// amdgpu-toolchain ships the TDM header and uses 6 args).
#if __has_builtin(__builtin_amdgcn_tensor_load_to_lds)
#define HAVE_TDM 1
#if __has_include(<hip/amd_detail/amd_gfx1250_TDM.h>)
#define TDM_ARGS6 1
#endif
#endif

union Frag16 {
    v16bf v;
    unsigned short u[16];
    uint4 q[2];
};

__device__ __forceinline__ unsigned short f2bf(float f) {
    unsigned int u = __float_as_uint(f);
    u += 0x7FFFu + ((u >> 16) & 1u);   // round-to-nearest-even
    return (unsigned short)(u >> 16);
}

// ---------------------------------------------------------------------------
// Stage 1: QKV projection.  C[16384,384] = X[16384,2048] * W[2048,384] + b
// bf16 WMMA with fp32 accumulation; Q gets 1/sqrt(128) baked in at store.
// Grid: (3 n-tiles of 128, 128 m-tiles of 128), block 256 (8 waves).
// ---------------------------------------------------------------------------
__launch_bounds__(256)
__global__ void qkv_proj_kernel(const float* __restrict__ x,
                                const float* __restrict__ W,
                                const float* __restrict__ bias,
                                unsigned short* __restrict__ qb,
                                unsigned short* __restrict__ kb,
                                unsigned short* __restrict__ vb)
{
    __shared__ unsigned short Xs[128][32];   // [m_local][k]   bf16
    __shared__ unsigned short WsT[128][32];  // [n_local][k]   bf16 (W transposed)

    const int tid   = threadIdx.x;
    const int wave  = tid >> 5;
    const int lane  = tid & 31;
    const int lhalf = lane & 15;
    const int hi    = lane >> 4;

    const int n0 = blockIdx.x * 128;
    const int m0 = blockIdx.y * 128;

    v8f zero = {};
    v8f acc[8];
#pragma unroll
    for (int t = 0; t < 8; ++t) acc[t] = zero;

    for (int ks = 0; ks < N_EMBD / 32; ++ks) {
        const int e0 = ks * 32;
        __syncthreads();
        // X tile: 128x32 fp32 -> bf16, coalesced float4 reads
#pragma unroll
        for (int i = 0; i < 4; ++i) {
            int off = (tid + 256 * i) * 4;
            int row = off >> 5;
            int e   = off & 31;
            const float4 f = *(const float4*)(x + (size_t)(m0 + row) * N_EMBD + e0 + e);
            uint2 pk;
            pk.x = (unsigned)f2bf(f.x) | ((unsigned)f2bf(f.y) << 16);
            pk.y = (unsigned)f2bf(f.z) | ((unsigned)f2bf(f.w) << 16);
            *(uint2*)&Xs[row][e] = pk;
        }
        // W tile: 32x128 fp32, stored transposed WsT[n][e]
#pragma unroll
        for (int i = 0; i < 4; ++i) {
            int off = (tid + 256 * i) * 4;
            int e   = off >> 7;
            int n   = off & 127;
            const float4 f = *(const float4*)(W + (size_t)(e0 + e) * QKV3 + n0 + n);
            WsT[n + 0][e] = f2bf(f.x);
            WsT[n + 1][e] = f2bf(f.y);
            WsT[n + 2][e] = f2bf(f.z);
            WsT[n + 3][e] = f2bf(f.w);
        }
        __syncthreads();

        // A fragment: 16x32, lane=row (m)
        Frag16 a;
        {
            const unsigned short* base = &Xs[wave * 16 + lhalf][hi * 8];
            a.q[0] = *(const uint4*)(base);
            a.q[1] = *(const uint4*)(base + 16);
        }
        // Preload B fragments in groups of 4, then issue the 4 WMMAs,
        // so DS latency overlaps the matrix pipe instead of dscnt-0 drains.
#pragma unroll
        for (int g = 0; g < 2; ++g) {
            Frag16 bf[4];
#pragma unroll
            for (int j = 0; j < 4; ++j) {
                const unsigned short* base = &WsT[(g * 4 + j) * 16 + lhalf][hi * 8];
                bf[j].q[0] = *(const uint4*)(base);
                bf[j].q[1] = *(const uint4*)(base + 16);
            }
#pragma unroll
            for (int j = 0; j < 4; ++j) {
                acc[g * 4 + j] = __builtin_amdgcn_wmma_f32_16x16x32_bf16(
                    false, a.v, false, bf[j].v, (short)0, acc[g * 4 + j], false, false);
            }
        }
    }

    // Epilogue: bias, split to Q/K/V (branch uniform per block: n0 in {0,128,256})
    const float scale = 0.08838834764831845f;  // 1/sqrt(128)
#pragma unroll
    for (int t = 0; t < 8; ++t) {
        const int n  = n0 + t * 16 + lhalf;
        const float bv = bias[n];
#pragma unroll
        for (int r = 0; r < 8; ++r) {
            const int m = m0 + r + 8 * hi;
            const float v = acc[t][r] + bv;
            if (n < HEADD)
                qb[(size_t)m * HEADD + n] = f2bf(v * scale);
            else if (n < 2 * HEADD)
                kb[(size_t)m * HEADD + (n - HEADD)] = f2bf(v);
            else
                vb[(size_t)m * HEADD + (n - 2 * HEADD)] = f2bf(v);
        }
    }
}

// ---------------------------------------------------------------------------
// Stage 2: flash attention. One 16-row query tile per wave, 8 waves/block.
// K tile staged into padded LDS by the Tensor Data Mover (pad_interval=64
// DWORDs, pad_amount=4 DWORDs reproduces Ks[32][136]); V transposed manually.
// Scores + PV via bf16 WMMA with fp32 online softmax.
// Grid: (SEQ/128, BATCH), block 256.
// ---------------------------------------------------------------------------
__launch_bounds__(256)
__global__ void flash_attn_kernel(const unsigned short* __restrict__ qb,
                                  const unsigned short* __restrict__ kb,
                                  const unsigned short* __restrict__ vb,
                                  float* __restrict__ out)
{
    __shared__ unsigned short Ks[32][136];   // [key][d], +8 pad: conflict-free b128
    __shared__ unsigned short Vt[128][48];   // [d][key], transposed V
    __shared__ unsigned short Pl[8][16][32]; // per-wave P transpose buffer

    const int tid   = threadIdx.x;
    const int wave  = tid >> 5;
    const int lane  = tid & 31;
    const int lhalf = lane & 15;
    const int hi    = lane >> 4;

    const int batch = blockIdx.y;
    const int q0    = blockIdx.x * 128 + wave * 16;
    const size_t rowbase = (size_t)batch * SEQ;

    // Q fragments for the whole pass: 4 x (16x32) over d=128
    Frag16 qf[4];
#pragma unroll
    for (int f = 0; f < 4; ++f) {
        const unsigned short* base =
            qb + (rowbase + q0 + lhalf) * HEADD + f * 32 + hi * 8;
        qf[f].q[0] = *(const uint4*)(base);
        qf[f].q[1] = *(const uint4*)(base + 16);
    }

    v8f zero = {};
    v8f O[8];
#pragma unroll
    for (int t = 0; t < 8; ++t) O[t] = zero;
    float mrow[8], lrow[8];
#pragma unroll
    for (int r = 0; r < 8; ++r) { mrow[r] = -1e30f; lrow[r] = 0.0f; }

    const unsigned short* Kg = kb + rowbase * HEADD;
    const unsigned short* Vg = vb + rowbase * HEADD;

#ifdef HAVE_TDM
    const unsigned ks_lds = (unsigned)(unsigned long long)(void*)&Ks[0][0];
#endif

    for (int kt = 0; kt < SEQ / 32; ++kt) {
        __syncthreads();

#ifdef HAVE_TDM
        // K tile 32x128 via Tensor Data Mover (one wave issues; EXEC ignored).
        if (wave == 0) {
            const unsigned long long ga =
                (unsigned long long)(Kg + (size_t)kt * 32 * HEADD);
            u32x4 g0;
            g0.x = 1u;                               // count=1, user descriptor
            g0.y = ks_lds;                           // lds_addr (bytes)
            g0.z = (unsigned)(ga & 0xFFFFFFFFull);   // global_addr[31:0]
            g0.w = (unsigned)((ga >> 32) & 0x1FFFFFFull) | 0x80000000u; // [56:32] | type=2
            i32x8 g1;
            g1[0] = 0x07510000;        // data_size=2B, pad_enable, interval=64dw, amount=4dw
            g1[1] = (int)(128u << 16); // tensor_dim0 = 128
            g1[2] = (int)(4096u << 16);// tensor_dim1 = 4096 (dim0 hi16 = 0)
            g1[3] = (int)(128u << 16); // tile_dim0 = 128 (dim1 hi16 = 0)
            g1[4] = 32;                // tile_dim1 = 32
            g1[5] = 128;               // tensor_dim0_stride = 128
            g1[6] = 0;
            g1[7] = 0;
            i32x4 z4 = {};
#ifdef TDM_ARGS6
            i32x8 z8 = {};
            __builtin_amdgcn_tensor_load_to_lds(g0, g1, z4, z4, z8, 0);
#else
            __builtin_amdgcn_tensor_load_to_lds(g0, g1, z4, z4, 0);
#endif
        }
#else
        // Fallback: manual K tile load 32x128 -> Ks[key][d]
#pragma unroll
        for (int i = 0; i < 4; ++i) {
            int off = (tid + 256 * i) * 4;
            int key = off >> 7;
            int d   = off & 127;
            uint2 pk = *(const uint2*)(Kg + (size_t)(kt * 32 + key) * HEADD + d);
            *(uint2*)&Ks[key][d] = pk;
        }
#endif

        // V tile 32x128 -> Vt[d][key] (transposed; overlaps the TDM DMA)
#pragma unroll
        for (int i = 0; i < 4; ++i) {
            int off = (tid + 256 * i) * 4;
            int key = off >> 7;
            int d   = off & 127;
            uint2 pk = *(const uint2*)(Vg + (size_t)(kt * 32 + key) * HEADD + d);
            Vt[d + 0][key] = (unsigned short)(pk.x & 0xFFFFu);
            Vt[d + 1][key] = (unsigned short)(pk.x >> 16);
            Vt[d + 2][key] = (unsigned short)(pk.y & 0xFFFFu);
            Vt[d + 3][key] = (unsigned short)(pk.y >> 16);
        }
        // prefetch next step's K/V tiles into cache (global_prefetch_b8)
        {
            int nkt = (kt + 1) & (SEQ / 32 - 1);
            __builtin_prefetch(Kg + (size_t)nkt * 32 * HEADD + tid * 16, 0, 3);
            __builtin_prefetch(Vg + (size_t)nkt * 32 * HEADD + tid * 16, 0, 3);
        }

#ifdef HAVE_TDM
        if (wave == 0) __builtin_amdgcn_s_wait_tensorcnt(0);
#endif
        __syncthreads();

        // Scores: S = Qscaled * K^T, two 16x16 tiles (key halves), K-dim = d
        v8f S0 = zero, S1 = zero;
#pragma unroll
        for (int f2 = 0; f2 < 2; ++f2) {
            Frag16 b[4];
#pragma unroll
            for (int j = 0; j < 2; ++j) {
                const int f = f2 * 2 + j;
                const unsigned short* k0 = &Ks[lhalf][f * 32 + hi * 8];
                const unsigned short* k1 = &Ks[16 + lhalf][f * 32 + hi * 8];
                b[2 * j + 0].q[0] = *(const uint4*)(k0);
                b[2 * j + 0].q[1] = *(const uint4*)(k0 + 16);
                b[2 * j + 1].q[0] = *(const uint4*)(k1);
                b[2 * j + 1].q[1] = *(const uint4*)(k1 + 16);
            }
#pragma unroll
            for (int j = 0; j < 2; ++j) {
                const int f = f2 * 2 + j;
                S0 = __builtin_amdgcn_wmma_f32_16x16x32_bf16(
                    false, qf[f].v, false, b[2 * j + 0].v, (short)0, S0, false, false);
                S1 = __builtin_amdgcn_wmma_f32_16x16x32_bf16(
                    false, qf[f].v, false, b[2 * j + 1].v, (short)0, S1, false, false);
            }
        }

        // Online softmax. D layout: lane=col(n), VGPR r = row r+8*hi.
        float p0[8], p1[8], corr[8];
#pragma unroll
        for (int r = 0; r < 8; ++r) {
            float mx = fmaxf(S0[r], S1[r]);
            mx = fmaxf(mx, __shfl_xor(mx, 1, 32));
            mx = fmaxf(mx, __shfl_xor(mx, 2, 32));
            mx = fmaxf(mx, __shfl_xor(mx, 4, 32));
            mx = fmaxf(mx, __shfl_xor(mx, 8, 32));
            float mnew = fmaxf(mrow[r], mx);
            corr[r] = __expf(mrow[r] - mnew);
            p0[r]   = __expf(S0[r] - mnew);
            p1[r]   = __expf(S1[r] - mnew);
            float s = p0[r] + p1[r];
            s += __shfl_xor(s, 1, 32);
            s += __shfl_xor(s, 2, 32);
            s += __shfl_xor(s, 4, 32);
            s += __shfl_xor(s, 8, 32);
            lrow[r] = lrow[r] * corr[r] + s;
            mrow[r] = mnew;
        }
#pragma unroll
        for (int t = 0; t < 8; ++t)
#pragma unroll
            for (int r = 0; r < 8; ++r)
                O[t][r] *= corr[r];

        // Transpose P (16 rows x 32 keys) through per-wave LDS into A layout.
        // Same-wave DS ops are in-order; s_wait_dscnt + memory clobber enforce it.
#pragma unroll
        for (int r = 0; r < 8; ++r) {
            Pl[wave][r + 8 * hi][lhalf]      = f2bf(p0[r]);
            Pl[wave][r + 8 * hi][16 + lhalf] = f2bf(p1[r]);
        }
        asm volatile("s_wait_dscnt 0" ::: "memory");

        Frag16 pf;
        {
            const unsigned short* base = &Pl[wave][lhalf][hi * 8];
            pf.q[0] = *(const uint4*)(base);
            pf.q[1] = *(const uint4*)(base + 16);
        }
        // O += P * V : preload 4 B fragments, then 4 WMMAs, twice
#pragma unroll
        for (int g = 0; g < 2; ++g) {
            Frag16 vf[4];
#pragma unroll
            for (int j = 0; j < 4; ++j) {
                const unsigned short* base = &Vt[(g * 4 + j) * 16 + lhalf][hi * 8];
                vf[j].q[0] = *(const uint4*)(base);
                vf[j].q[1] = *(const uint4*)(base + 16);
            }
#pragma unroll
            for (int j = 0; j < 4; ++j) {
                O[g * 4 + j] = __builtin_amdgcn_wmma_f32_16x16x32_bf16(
                    false, pf.v, false, vf[j].v, (short)0, O[g * 4 + j], false, false);
            }
        }
    }

    // Normalize and store fp32 output
#pragma unroll
    for (int t = 0; t < 8; ++t) {
        const int d = t * 16 + lhalf;
#pragma unroll
        for (int r = 0; r < 8; ++r) {
            const float inv = 1.0f / lrow[r];
            const int row = q0 + r + 8 * hi;
            out[((size_t)batch * SEQ + row) * HEADD + d] = O[t][r] * inv;
        }
    }
}

// ---------------------------------------------------------------------------
extern "C" void kernel_launch(void* const* d_in, const int* in_sizes, int n_in,
                              void* d_out, int out_size, void* d_ws, size_t ws_size,
                              hipStream_t stream) {
    const float* x  = (const float*)d_in[0];
    const float* W  = (const float*)d_in[1];
    const float* b  = (const float*)d_in[2];
    float* out      = (float*)d_out;

    const size_t plane = (size_t)BATCH * SEQ * HEADD;   // elements per Q/K/V plane
    unsigned short* qb = (unsigned short*)d_ws;
    unsigned short* kb = qb + plane;
    unsigned short* vb = kb + plane;

    qkv_proj_kernel<<<dim3(QKV3 / 128, (BATCH * SEQ) / 128), 256, 0, stream>>>(
        x, W, b, qb, kb, vb);
    flash_attn_kernel<<<dim3(SEQ / 128, BATCH), 256, 0, stream>>>(
        qb, kb, vb, out);
}
